// KhanhLLM_59528246722653
// MI455X (gfx1250) — matrix operand order repaired and checked by reading, not compiled, not run on hardware
//
#include <hip/hip_runtime.h>

// ---------------------------------------------------------------------------
// Model dims (compile-time constants from the reference)
// ---------------------------------------------------------------------------
#define DD   1024            // model dim
#define FF   4096            // ffn dim
#define EE   8               // experts
#define KK_TOP 2             // top-k
#define HH   16              // heads
#define LL   14              // layers
#define VV   50000           // vocab
#define BB   2
#define SS   512
#define NT   (BB*SS)         // 1024 tokens
#define HDIM 64
#define MOE_CAP 2560         // 2048 slots + per-expert 64-padding slack

// GEMM tiling
#define BM 64
#define BN 128
#define BK2 64               // k-depth per LDS stage (2 WMMA k-steps)
#define LDSPAD 8             // bf16 pad -> 144B row stride (16B aligned)

typedef __attribute__((ext_vector_type(16))) __bf16         v16bf;
typedef __attribute__((ext_vector_type(4)))  __bf16         bf4;
typedef __attribute__((ext_vector_type(8)))  float          v8f;
typedef __attribute__((ext_vector_type(8)))  unsigned short us8;
typedef __attribute__((ext_vector_type(16))) unsigned short us16;

__device__ __forceinline__ bf4 f4_to_bf4(float4 v) {
    bf4 r;
    r.x = (__bf16)v.x; r.y = (__bf16)v.y; r.z = (__bf16)v.z; r.w = (__bf16)v.w;
    return r;
}

// A-fragment (16x32 bf16): lane<16 row=lane K{0..7,16..23}; lane>=16 +8
__device__ __forceinline__ v16bf ldsA_frag(const __bf16* p) {
    us16 u;
    ((us8*)&u)[0] = *(const us8*)p;
    ((us8*)&u)[1] = *(const us8*)(p + 16);
    return __builtin_bit_cast(v16bf, u);
}
// B-fragment (32x16 bf16): lane<16 col=lane K{0..7,16..23}; lane>=16 +8
__device__ __forceinline__ v16bf ldsB_frag(const __bf16* p) {
    us16 u;
    ((us8*)&u)[0] = *(const us8*)p;
    ((us8*)&u)[1] = *(const us8*)(p + 16);
    return __builtin_bit_cast(v16bf, u);
}

// ---------------------------------------------------------------------------
// WMMA GEMM:  C[M,N] = act( A[M,K] * B + bias )   (f32 in/out, bf16 WMMA core)
//  BT:   0 -> B is [K,N] row-major ; 1 -> B is [N,K] row-major (i.e. x @ W^T)
//  ACT:  0 -> none ; 1 -> exact GELU
//  MODE: 0 -> plain ; 1 -> MoE gather ; 2 -> MoE scatter-add
// Double-buffered LDS pipeline, one barrier per 64-deep k-stage (8 WMMAs),
// plus 2-stages-ahead global_prefetch of the weight stream.
// ---------------------------------------------------------------------------
template <int BT, int ACT, int MODE>
__global__ void __launch_bounds__(256)
wmma_gemm(const float* __restrict__ A, int lda,
          const float* __restrict__ Bw, long b_stride_e,
          const float* __restrict__ bias, int bias_stride_e,
          float* __restrict__ C, int ldc,
          int M, int N, int Kd,
          const int*   __restrict__ row_idx,
          const float* __restrict__ row_w,
          const int*   __restrict__ padded_off)
{
    __shared__ __bf16 As[2][BM][BK2 + LDSPAD];
    __shared__ __bf16 Bs[2][BN][BK2 + LDSPAD];
    __shared__ int   orow_s[BM];
    __shared__ float rw_s[BM];
    __shared__ int   arow_s[BM];

    const int tid    = threadIdx.x;
    const int tile_n = blockIdx.x * BN;

    const float* Bp    = Bw;
    const float* biasp = bias;

    if (MODE == 0) {
        int tile_m = blockIdx.y * BM;
        if (tid < BM) {
            int gm = tile_m + tid;
            int ok = (gm < M);
            arow_s[tid] = ok ? gm : -1;
            orow_s[tid] = ok ? gm : -1;
            rw_s[tid]   = 1.f;
        }
    } else {
        int slot0 = blockIdx.y * BM;
        if (slot0 >= padded_off[EE]) return;     // uniform exit, before barrier
        int e = 0;
        while (e < EE - 1 && slot0 >= padded_off[e + 1]) e++;
        Bp    = Bw   + (long)e * b_stride_e;
        biasp = bias + (long)e * bias_stride_e;
        if (tid < BM) {
            int slot = slot0 + tid;
            int r    = row_idx[slot];            // -1 on padding slots
            if (MODE == 1) {
                arow_s[tid] = r;
                orow_s[tid] = (r >= 0) ? slot : -1;
                rw_s[tid]   = 1.f;
            } else {
                arow_s[tid] = (r >= 0) ? slot : -1;
                orow_s[tid] = r;
                rw_s[tid]   = row_w[slot];
            }
        }
    }
    __syncthreads();

    // ---- hoisted per-thread staging pointers (loop-invariant) ----
    const int kcA = (tid & 15) * 4;
    const float* aPtr[4];
    #pragma unroll
    for (int t = 0; t < 4; t++) {
        int m = (tid >> 4) + t * 16;
        int r = arow_s[m];
        aPtr[t] = (r >= 0) ? (A + (size_t)r * lda + kcA) : nullptr;
    }
    const int kcB  = (tid & 15) * 4;           // BT=1: k-chunk
    const int colB = tile_n + (tid & 31) * 4;  // BT=0: fixed column group
    const float* bPtr[8];
    if (BT) {
        #pragma unroll
        for (int t = 0; t < 8; t++) {
            int n   = (tid >> 4) + t * 16;
            int col = tile_n + n;
            bPtr[t] = (col < N) ? (Bp + (size_t)col * Kd + kcB) : nullptr;
        }
    }

    const int wid  = tid >> 5;
    const int lane = tid & 31;
    const int wm   = wid & 1;        // 2 waves over M
    const int wn   = wid >> 1;       // 4 waves over N
    const int rsel  = lane & 15;
    const int rhalf = lane >> 4;

    v8f acc[2][2] = {};
    float4 aReg[4];
    float4 bReg[8];

    const int kTiles = Kd / BK2;

    auto LOAD = [&](int k0) {
        #pragma unroll
        for (int t = 0; t < 4; t++) {
            aReg[t] = aPtr[t] ? *(const float4*)(aPtr[t] + k0)
                              : make_float4(0.f, 0.f, 0.f, 0.f);
        }
        if (BT) {
            #pragma unroll
            for (int t = 0; t < 8; t++) {
                bReg[t] = bPtr[t] ? *(const float4*)(bPtr[t] + k0)
                                  : make_float4(0.f, 0.f, 0.f, 0.f);
            }
        } else {
            #pragma unroll
            for (int t = 0; t < 8; t++) {
                int kk = k0 + (tid >> 5) + t * 8;
                if (colB + 3 < N) {
                    bReg[t] = *(const float4*)(Bp + (size_t)kk * N + colB);
                } else {
                    float tmp[4] = {0.f, 0.f, 0.f, 0.f};
                    for (int j2 = 0; j2 < 4; j2++)
                        if (colB + j2 < N) tmp[j2] = Bp[(size_t)kk * N + colB + j2];
                    bReg[t] = make_float4(tmp[0], tmp[1], tmp[2], tmp[3]);
                }
            }
        }
    };
    // speculative prefetch of the stage AFTER next (weight stream + A rows)
    auto PREFETCH = [&](int k0) {
        if (k0 < Kd) {
            if (BT) {
                if (bPtr[0]) __builtin_prefetch(bPtr[0] + k0, 0, 1);
                if (bPtr[4]) __builtin_prefetch(bPtr[4] + k0, 0, 1);
            } else {
                int kk = k0 + (tid >> 5);
                __builtin_prefetch(Bp + (size_t)kk * N + colB, 0, 1);
                __builtin_prefetch(Bp + (size_t)(kk + 32) * N + colB, 0, 1);
            }
            if (aPtr[0]) __builtin_prefetch(aPtr[0] + k0, 0, 1);
        }
    };
    auto STORE = [&](int buf) {
        #pragma unroll
        for (int t = 0; t < 4; t++) {
            int m = (tid >> 4) + t * 16;
            *(bf4*)&As[buf][m][kcA] = f4_to_bf4(aReg[t]);
        }
        if (BT) {
            #pragma unroll
            for (int t = 0; t < 8; t++) {
                int n = (tid >> 4) + t * 16;
                *(bf4*)&Bs[buf][n][kcB] = f4_to_bf4(bReg[t]);
            }
        } else {
            #pragma unroll
            for (int t = 0; t < 8; t++) {
                int kk = (tid >> 5) + t * 8;
                int nc = (tid & 31) * 4;
                Bs[buf][nc + 0][kk] = (__bf16)bReg[t].x;
                Bs[buf][nc + 1][kk] = (__bf16)bReg[t].y;
                Bs[buf][nc + 2][kk] = (__bf16)bReg[t].z;
                Bs[buf][nc + 3][kk] = (__bf16)bReg[t].w;
            }
        }
    };
    auto COMPUTE = [&](int buf) {
        #pragma unroll
        for (int ks = 0; ks < 2; ks++) {
            v16bf afr[2], bfr[2];
            #pragma unroll
            for (int sm = 0; sm < 2; sm++)
                afr[sm] = ldsA_frag(&As[buf][wm * 32 + sm * 16 + rsel][ks * 32 + rhalf * 8]);
            #pragma unroll
            for (int sn = 0; sn < 2; sn++)
                bfr[sn] = ldsB_frag(&Bs[buf][wn * 32 + sn * 16 + rsel][ks * 32 + rhalf * 8]);
            #pragma unroll
            for (int sm = 0; sm < 2; sm++)
                #pragma unroll
                for (int sn = 0; sn < 2; sn++)
                    acc[sm][sn] = __builtin_amdgcn_wmma_f32_16x16x32_bf16(
                        false, afr[sm], false, bfr[sn],
                        (short)0, acc[sm][sn], false, false);
        }
    };

    LOAD(0);
    PREFETCH(BK2);
    STORE(0);
    __syncthreads();
    for (int kt = 0; kt < kTiles; kt++) {
        const int nxt = kt + 1;
        if (nxt < kTiles) {
            LOAD(nxt * BK2);                   // issue global loads early
            PREFETCH((kt + 2) * BK2);          // warm L2/L0 two stages ahead
        }
        COMPUTE(kt & 1);                       // WMMAs hide the load latency
        if (nxt < kTiles) {
            STORE(nxt & 1);
            __syncthreads();
        }
    }

    // ---- epilogue ----
    #pragma unroll
    for (int sn = 0; sn < 2; sn++) {
        int col = tile_n + wn * 32 + sn * 16 + rsel;
        if (col >= N) continue;
        float bv = biasp[col];
        #pragma unroll
        for (int sm = 0; sm < 2; sm++) {
            int ml0 = wm * 32 + sm * 16 + rhalf * 8;
            #pragma unroll
            for (int v = 0; v < 8; v++) {
                int ml   = ml0 + v;
                int orow = orow_s[ml];
                if (orow < 0) continue;
                float val = acc[sm][sn][v] + bv;
                if (ACT == 1) val = 0.5f * val * (1.f + erff(val * 0.70710678118f));
                if (MODE == 2)
                    atomicAdd(&C[(size_t)orow * ldc + col], rw_s[ml] * val);
                else
                    C[(size_t)orow * ldc + col] = val;
            }
        }
    }
}

// ---------------------------------------------------------------------------
// Embedding:  x[n] = tok_emb[tokens[n]] + pos_emb[n % S]
// ---------------------------------------------------------------------------
__global__ void embed_kernel(const int* __restrict__ tokens,
                             const float* __restrict__ tok_emb,
                             const float* __restrict__ pos_emb,
                             float* __restrict__ x)
{
    int n  = blockIdx.x;
    int tk = tokens[n];
    int s  = n % SS;
    const float* te = tok_emb + (size_t)tk * DD;
    const float* pe = pos_emb + (size_t)s * DD;
    float* xr = x + (size_t)n * DD;
    for (int d = threadIdx.x; d < DD; d += 256) xr[d] = te[d] + pe[d];
}

// ---------------------------------------------------------------------------
// Flash attention with WMMA. Block = 128 threads (4 waves) per
// (64-query tile, b*h). Each wave owns 16 query rows -> softmax stats stay
// inside the wave (half-wave shfl reductions over the C-fragment columns).
// qkv row layout [q | k | v] of 3*D; Q pre-scaled by 1/sqrt(64).
// ---------------------------------------------------------------------------
__global__ void __launch_bounds__(128)
attn_wmma_kernel(const float* __restrict__ qkv, float* __restrict__ o)
{
    __shared__ __bf16 Qs[64][72];
    __shared__ __bf16 Ks[64][72];
    __shared__ __bf16 Vt[64][72];   // V transposed: Vt[d][key]
    __shared__ __bf16 Ps[64][72];

    const int tid   = threadIdx.x;
    const int wid   = tid >> 5;          // 4 waves, wave owns rows wid*16..+16
    const int lane  = tid & 31;
    const int rsel  = lane & 15;
    const int rhalf = lane >> 4;

    const int qb = blockIdx.x * 64;      // query tile start
    const int bh = blockIdx.y;
    const int h  = bh % HH;
    const int b  = bh / HH;

    const float* base = qkv + (size_t)(b * SS) * (3 * DD) + h * HDIM;

    // stage Q tile (scaled)
    #pragma unroll
    for (int t = 0; t < 8; t++) {
        int idx = tid + t * 128;
        int r   = idx >> 4;
        int c4  = (idx & 15) * 4;
        float4 v = *(const float4*)(base + (size_t)(qb + r) * (3 * DD) + c4);
        bf4 hv;
        hv.x = (__bf16)(v.x * 0.125f); hv.y = (__bf16)(v.y * 0.125f);
        hv.z = (__bf16)(v.z * 0.125f); hv.w = (__bf16)(v.w * 0.125f);
        *(bf4*)&Qs[r][c4] = hv;
    }
    __syncthreads();

    v8f acc_o[4] = {};
    float mrow[8], lrow[8];
    #pragma unroll
    for (int v = 0; v < 8; v++) { mrow[v] = -1e30f; lrow[v] = 0.f; }

    const int ktiles = qb / 64 + 1;      // causal: tiles fully above diag skipped
    for (int kt = 0; kt < ktiles; kt++) {
        const int k0 = kt * 64;

        // stage K (row-major = B-frag layout) and V transposed
        #pragma unroll
        for (int t = 0; t < 8; t++) {
            int idx = tid + t * 128;
            int r   = idx >> 4;
            int c4  = (idx & 15) * 4;
            float4 kv = *(const float4*)(base + (size_t)(k0 + r) * (3 * DD) + DD + c4);
            *(bf4*)&Ks[r][c4] = f4_to_bf4(kv);
            float4 vv = *(const float4*)(base + (size_t)(k0 + r) * (3 * DD) + 2 * DD + c4);
            Vt[c4 + 0][r] = (__bf16)vv.x;
            Vt[c4 + 1][r] = (__bf16)vv.y;
            Vt[c4 + 2][r] = (__bf16)vv.z;
            Vt[c4 + 3][r] = (__bf16)vv.w;
            // prefetch next key tile (speculative; dropped at last tile)
            if (kt + 1 < ktiles) {
                __builtin_prefetch(base + (size_t)(k0 + 64 + r) * (3 * DD) + DD + c4, 0, 1);
                __builtin_prefetch(base + (size_t)(k0 + 64 + r) * (3 * DD) + 2 * DD + c4, 0, 1);
            }
        }
        __syncthreads();

        // S = Q * K^T for this wave's 16 rows (4 col subtiles, k=64)
        v8f s[4] = {};
        #pragma unroll
        for (int ks = 0; ks < 2; ks++) {
            v16bf aq = ldsA_frag(&Qs[wid * 16 + rsel][ks * 32 + rhalf * 8]);
            #pragma unroll
            for (int sn = 0; sn < 4; sn++) {
                v16bf bk = ldsB_frag(&Ks[sn * 16 + rsel][ks * 32 + rhalf * 8]);
                s[sn] = __builtin_amdgcn_wmma_f32_16x16x32_bf16(
                    false, aq, false, bk, (short)0, s[sn], false, false);
            }
        }

        // causal mask + online softmax (rows rhalf*8+v, cols sn*16+rsel)
        const int rowg0 = qb + wid * 16 + rhalf * 8;
        float pv[4][8];
        float mtile[8];
        #pragma unroll
        for (int v = 0; v < 8; v++) mtile[v] = -1e30f;
        #pragma unroll
        for (int sn = 0; sn < 4; sn++) {
            int colg = k0 + sn * 16 + rsel;
            #pragma unroll
            for (int v = 0; v < 8; v++) {
                float sv = s[sn][v];
                if (colg > rowg0 + v) sv = -1e30f;   // causal
                pv[sn][v] = sv;
                mtile[v] = fmaxf(mtile[v], sv);
            }
        }
        #pragma unroll
        for (int v = 0; v < 8; v++) {
            float t = mtile[v];
            t = fmaxf(t, __shfl_xor(t, 1, 32));
            t = fmaxf(t, __shfl_xor(t, 2, 32));
            t = fmaxf(t, __shfl_xor(t, 4, 32));
            t = fmaxf(t, __shfl_xor(t, 8, 32));
            mtile[v] = t;
        }
        float rsum[8];
        #pragma unroll
        for (int v = 0; v < 8; v++) {
            float mnew = fmaxf(mrow[v], mtile[v]);
            float scl  = __expf(mrow[v] - mnew);
            mrow[v] = mnew;
            lrow[v] *= scl;
            #pragma unroll
            for (int sd = 0; sd < 4; sd++) acc_o[sd][v] *= scl;
            float rs = 0.f;
            #pragma unroll
            for (int sn = 0; sn < 4; sn++) {
                float p = __expf(pv[sn][v] - mnew);
                pv[sn][v] = p;
                rs += p;
            }
            rsum[v] = rs;
        }
        #pragma unroll
        for (int v = 0; v < 8; v++) {
            float t = rsum[v];
            t += __shfl_xor(t, 1, 32);
            t += __shfl_xor(t, 2, 32);
            t += __shfl_xor(t, 4, 32);
            t += __shfl_xor(t, 8, 32);
            lrow[v] += t;
        }
        // P -> LDS in A-fragment (row-major) layout
        #pragma unroll
        for (int sn = 0; sn < 4; sn++)
            #pragma unroll
            for (int v = 0; v < 8; v++)
                Ps[wid * 16 + rhalf * 8 + v][sn * 16 + rsel] = (__bf16)pv[sn][v];
        __syncthreads();

        // O += P @ V   (B = Vt[d][key])
        #pragma unroll
        for (int ks = 0; ks < 2; ks++) {
            v16bf ap = ldsA_frag(&Ps[wid * 16 + rsel][ks * 32 + rhalf * 8]);
            #pragma unroll
            for (int sd = 0; sd < 4; sd++) {
                v16bf bv = ldsB_frag(&Vt[sd * 16 + rsel][ks * 32 + rhalf * 8]);
                acc_o[sd] = __builtin_amdgcn_wmma_f32_16x16x32_bf16(
                    false, ap, false, bv, (short)0, acc_o[sd], false, false);
            }
        }
        __syncthreads();
    }

    // writeback: out[b, row, h*64 + d] = O / l
    #pragma unroll
    for (int sd = 0; sd < 4; sd++) {
        int col = h * HDIM + sd * 16 + rsel;
        #pragma unroll
        for (int v = 0; v < 8; v++) {
            int rowg = qb + wid * 16 + rhalf * 8 + v;
            o[(size_t)(b * SS + rowg) * DD + col] = acc_o[sd][v] / lrow[v];
        }
    }
}

// ---------------------------------------------------------------------------
// x = LayerNorm(x + r) * g + b   (in place over x), one block per row
// ---------------------------------------------------------------------------
__device__ __forceinline__ float block_sum_1024(float s, float* sm8)
{
    for (int off = 16; off; off >>= 1) s += __shfl_xor(s, off, 32);
    int wid = threadIdx.x >> 5, lane = threadIdx.x & 31;
    if (lane == 0) sm8[wid] = s;
    __syncthreads();
    float t = (threadIdx.x < 8) ? sm8[threadIdx.x] : 0.f;
    if (wid == 0)
        for (int off = 4; off; off >>= 1) t += __shfl_xor(t, off, 32);
    if (threadIdx.x == 0) sm8[0] = t;
    __syncthreads();
    float r = sm8[0];
    __syncthreads();
    return r;
}

__global__ void __launch_bounds__(256)
add_ln_kernel(float* __restrict__ x, const float* __restrict__ r,
              const float* __restrict__ g, const float* __restrict__ bt)
{
    __shared__ float sm8[8];
    const int row = blockIdx.x;
    float* xr = x + (size_t)row * DD;
    const float* rr = r + (size_t)row * DD;

    float v[4];
    float s = 0.f;
    #pragma unroll
    for (int i = 0; i < 4; i++) {
        int d = threadIdx.x + i * 256;
        v[i] = xr[d] + rr[d];
        s += v[i];
    }
    float mean = block_sum_1024(s, sm8) * (1.f / DD);

    float s2 = 0.f;
    #pragma unroll
    for (int i = 0; i < 4; i++) { float dd = v[i] - mean; s2 += dd * dd; }
    float var = block_sum_1024(s2, sm8) * (1.f / DD);
    float inv = rsqrtf(var + 1e-5f);

    #pragma unroll
    for (int i = 0; i < 4; i++) {
        int d = threadIdx.x + i * 256;
        xr[d] = (v[i] - mean) * inv * g[d] + bt[d];
    }
}

// ---------------------------------------------------------------------------
// MoE gate: one wave per token. Softmax over E=8, top-2, aux stats.
// ---------------------------------------------------------------------------
__global__ void __launch_bounds__(256)
gate_topk_kernel(const float* __restrict__ x,
                 const float* __restrict__ gw, const float* __restrict__ gb,
                 int* __restrict__ tok_sel, float* __restrict__ tok_w,
                 float* __restrict__ probs_sum, float* __restrict__ top1c,
                 int* __restrict__ exp_cnt)
{
    const int wid  = threadIdx.x >> 5;
    const int lane = threadIdx.x & 31;
    const int n    = blockIdx.x * 8 + wid;
    const float* xr = x + (size_t)n * DD;

    float g[EE];
    #pragma unroll
    for (int e = 0; e < EE; e++) {
        const float* wr = gw + (size_t)e * DD;
        float a = 0.f;
        for (int d = lane; d < DD; d += 32) a += xr[d] * wr[d];
        for (int off = 16; off; off >>= 1) a += __shfl_xor(a, off, 32);
        g[e] = a + gb[e];
    }
    float m = g[0];
    #pragma unroll
    for (int e = 1; e < EE; e++) m = fmaxf(m, g[e]);
    float p[EE], se = 0.f;
    #pragma unroll
    for (int e = 0; e < EE; e++) { p[e] = __expf(g[e] - m); se += p[e]; }
    float ise = 1.f / se;
    #pragma unroll
    for (int e = 0; e < EE; e++) p[e] *= ise;

    int i0 = 0;
    #pragma unroll
    for (int e = 1; e < EE; e++) if (p[e] > p[i0]) i0 = e;
    int i1 = (i0 == 0) ? 1 : 0;
    #pragma unroll
    for (int e = 0; e < EE; e++) if (e != i0 && p[e] > p[i1]) i1 = e;

    float w0 = p[i0], w1 = p[i1];
    float ws = w0 + w1;
    w0 /= ws; w1 /= ws;

    if (lane < EE) atomicAdd(&probs_sum[lane], p[lane]);
    if (lane == 0) {
        atomicAdd(&top1c[i0], 1.0f);
        tok_sel[n * 2 + 0] = i0;
        tok_sel[n * 2 + 1] = i1;
        tok_w[n * 2 + 0] = w0;
        tok_w[n * 2 + 1] = w1;
        atomicAdd(&exp_cnt[i0], 1);
        atomicAdd(&exp_cnt[i1], 1);
    }
}

__global__ void moe_scan_kernel(const int* __restrict__ exp_cnt,
                                int* __restrict__ padded_off)
{
    if (threadIdx.x == 0 && blockIdx.x == 0) {
        int off = 0;
        for (int e = 0; e < EE; e++) {
            padded_off[e] = off;
            off += (exp_cnt[e] + BM - 1) & ~(BM - 1);
        }
        padded_off[EE] = off;
    }
}

__global__ void moe_scatter_kernel(const int* __restrict__ tok_sel,
                                   const float* __restrict__ tok_w,
                                   const int* __restrict__ padded_off,
                                   int* __restrict__ cursor,
                                   int* __restrict__ row_idx,
                                   float* __restrict__ row_w)
{
    int n = blockIdx.x * 256 + threadIdx.x;
    if (n >= NT) return;
    #pragma unroll
    for (int k = 0; k < KK_TOP; k++) {
        int e   = tok_sel[n * 2 + k];
        int pos = atomicAdd(&cursor[e], 1);
        int s   = padded_off[e] + pos;
        row_idx[s] = n;
        row_w[s]   = tok_w[n * 2 + k];
    }
}

__global__ void aux_kernel(const float* __restrict__ probs_sum,
                           const float* __restrict__ top1c,
                           float* __restrict__ aux_acc)
{
    if (threadIdx.x == 0 && blockIdx.x == 0) {
        float a = 0.f;
        for (int e = 0; e < EE; e++) a += probs_sum[e] * top1c[e];
        aux_acc[0] += (float)EE * a / ((float)NT * (float)NT);
    }
}

__global__ void fill_f32(float* p, float v, int count)
{
    int i = blockIdx.x * 256 + threadIdx.x;
    if (i < count) p[i] = v;
}
__global__ void fill_i32(int* p, int v, int count)
{
    int i = blockIdx.x * 256 + threadIdx.x;
    if (i < count) p[i] = v;
}
__global__ void store_aux_kernel(const float* __restrict__ aux_acc,
                                 float* __restrict__ dst)
{
    if (threadIdx.x == 0 && blockIdx.x == 0) dst[0] = aux_acc[0];
}

// ---------------------------------------------------------------------------
// kernel_launch
// ---------------------------------------------------------------------------
extern "C" void kernel_launch(void* const* d_in, const int* in_sizes, int n_in,
                              void* d_out, int out_size, void* d_ws, size_t ws_size,
                              hipStream_t stream)
{
    (void)in_sizes; (void)n_in; (void)out_size; (void)ws_size;

    const int*   tokens     = (const int*)  d_in[0];
    const float* tok_emb    = (const float*)d_in[1];
    const float* pos_emb    = (const float*)d_in[2];
    const float* in_proj_w  = (const float*)d_in[3];
    const float* in_proj_b  = (const float*)d_in[4];
    const float* attn_out_w = (const float*)d_in[5];
    const float* attn_out_b = (const float*)d_in[6];
    const float* ln1_g      = (const float*)d_in[7];
    const float* ln1_b      = (const float*)d_in[8];
    const float* ln2_g      = (const float*)d_in[9];
    const float* ln2_b      = (const float*)d_in[10];
    const float* dense_w1   = (const float*)d_in[11];
    const float* dense_b1   = (const float*)d_in[12];
    const float* dense_w2   = (const float*)d_in[13];
    const float* dense_b2   = (const float*)d_in[14];
    const float* gate_w     = (const float*)d_in[15];
    const float* gate_b     = (const float*)d_in[16];
    const float* moe_w1     = (const float*)d_in[17];
    const float* moe_b1     = (const float*)d_in[18];
    const float* moe_w2     = (const float*)d_in[19];
    const float* moe_b2     = (const float*)d_in[20];
    const float* out_w      = (const float*)d_in[21];
    const float* out_b      = (const float*)d_in[22];

    float* logits = (float*)d_out;

    // workspace layout
    float* w = (float*)d_ws;
    float* x      = w;  w += (size_t)NT * DD;
    float* qkv    = w;  w += (size_t)NT * 3 * DD;
    float* attn_o = w;  w += (size_t)NT * DD;
    float* abuf   = w;  w += (size_t)NT * DD;
    float* hbuf   = w;  w += (size_t)NT * FF;
    float* hmoe   = w;  w += (size_t)MOE_CAP * FF;
    float* fbuf   = w;  w += (size_t)NT * DD;
    float* probs_sum = w; w += EE;
    float* top1c     = w; w += EE;
    float* aux_acc   = w; w += 1;
    float* tok_w     = w; w += NT * 2;
    float* row_w     = w; w += MOE_CAP;
    int* ib = (int*)w;
    int* exp_cnt    = ib; ib += EE;
    int* cursor     = ib; ib += EE;
    int* padded_off = ib; ib += EE + 1;
    int* tok_sel    = ib; ib += NT * 2;
    int* row_idx    = ib; ib += MOE_CAP;

    const dim3 blk(256);

    fill_f32<<<1, blk, 0, stream>>>(aux_acc, 0.f, 1);
    embed_kernel<<<NT, blk, 0, stream>>>(tokens, tok_emb, pos_emb, x);

    for (int i = 0; i < LL; i++) {
        const int j = i / 2;

        // QKV projection: x @ in_proj_w[i]^T + b  -> qkv [NT, 3D]
        wmma_gemm<1, 0, 0><<<dim3(3 * DD / BN, NT / BM), blk, 0, stream>>>(
            x, DD, in_proj_w + (size_t)i * 3 * DD * DD, 0,
            in_proj_b + (size_t)i * 3 * DD, 0,
            qkv, 3 * DD, NT, 3 * DD, DD, nullptr, nullptr, nullptr);

        // flash attention (WMMA)
        attn_wmma_kernel<<<dim3(SS / 64, BB * HH), dim3(128), 0, stream>>>(qkv, attn_o);

        // out projection: attn_o @ attn_out_w[i]^T + b -> abuf
        wmma_gemm<1, 0, 0><<<dim3(DD / BN, NT / BM), blk, 0, stream>>>(
            attn_o, DD, attn_out_w + (size_t)i * DD * DD, 0,
            attn_out_b + (size_t)i * DD, 0,
            abuf, DD, NT, DD, DD, nullptr, nullptr, nullptr);

        add_ln_kernel<<<NT, blk, 0, stream>>>(
            x, abuf, ln1_g + (size_t)i * DD, ln1_b + (size_t)i * DD);

        if ((i & 1) == 0) {
            // dense FFN
            wmma_gemm<0, 1, 0><<<dim3(FF / BN, NT / BM), blk, 0, stream>>>(
                x, DD, dense_w1 + (size_t)j * DD * FF, 0,
                dense_b1 + (size_t)j * FF, 0,
                hbuf, FF, NT, FF, DD, nullptr, nullptr, nullptr);
            wmma_gemm<0, 0, 0><<<dim3(DD / BN, NT / BM), blk, 0, stream>>>(
                hbuf, FF, dense_w2 + (size_t)j * FF * DD, 0,
                dense_b2 + (size_t)j * DD, 0,
                fbuf, DD, NT, DD, FF, nullptr, nullptr, nullptr);
        } else {
            // MoE FFN with top-2 gathered grouped GEMM
            fill_f32<<<1, blk, 0, stream>>>(probs_sum, 0.f, 2 * EE);       // probs_sum + top1c
            fill_i32<<<1, blk, 0, stream>>>(exp_cnt, 0, 2 * EE);           // exp_cnt + cursor
            gate_topk_kernel<<<NT / 8, blk, 0, stream>>>(
                x, gate_w + (size_t)j * EE * DD, gate_b + (size_t)j * EE,
                tok_sel, tok_w, probs_sum, top1c, exp_cnt);
            moe_scan_kernel<<<1, 1, 0, stream>>>(exp_cnt, padded_off);
            fill_i32<<<(MOE_CAP + 255) / 256, blk, 0, stream>>>(row_idx, -1, MOE_CAP);
            moe_scatter_kernel<<<(NT + 255) / 256, blk, 0, stream>>>(
                tok_sel, tok_w, padded_off, cursor, row_idx, row_w);
            aux_kernel<<<1, 1, 0, stream>>>(probs_sum, top1c, aux_acc);

            fill_f32<<<(NT * DD) / 256, blk, 0, stream>>>(fbuf, 0.f, NT * DD);

            // expert GEMM1: h = gelu( gather(x) @ w1[e] + b1[e] )
            wmma_gemm<0, 1, 1><<<dim3(FF / BN, MOE_CAP / BM), blk, 0, stream>>>(
                x, DD, moe_w1 + (size_t)j * EE * DD * FF, (long)DD * FF,
                moe_b1 + (size_t)j * EE * FF, FF,
                hmoe, FF, NT, FF, DD, row_idx, row_w, padded_off);
            // expert GEMM2: fbuf[token] += w_slot * ( h @ w2[e] + b2[e] )
            wmma_gemm<0, 0, 2><<<dim3(DD / BN, MOE_CAP / BM), blk, 0, stream>>>(
                hmoe, FF, moe_w2 + (size_t)j * EE * FF * DD, (long)FF * DD,
                moe_b2 + (size_t)j * EE * DD, DD,
                fbuf, DD, MOE_CAP, DD, FF, row_idx, row_w, padded_off);
        }

        add_ln_kernel<<<NT, blk, 0, stream>>>(
            x, fbuf, ln2_g + (size_t)i * DD, ln2_b + (size_t)i * DD);
    }

    // logits = x @ out_w^T + out_b   [NT, V]
    wmma_gemm<1, 0, 0><<<dim3((VV + BN - 1) / BN, NT / BM), blk, 0, stream>>>(
        x, DD, out_w, 0, out_b, 0,
        logits, VV, NT, VV, DD, nullptr, nullptr, nullptr);

    store_aux_kernel<<<1, 1, 0, stream>>>(aux_acc, logits + (size_t)NT * VV);
}
